// FusedBitLinear_41815801594117
// MI455X (gfx1250) — 8-bit matrix operand order repaired and checked by reading, compiled, not run on hardware
//
#include <hip/hip_runtime.h>
#include <math.h>


typedef int v8i __attribute__((ext_vector_type(8)));

#define D_IN   4096
#define EPS_LN 1e-5f
#define EPS_Q  1e-5f
#define QB     127.0f

__device__ __forceinline__ float waveSum(float v) {
#pragma unroll
  for (int off = 16; off > 0; off >>= 1) v += __shfl_xor(v, off, 32);
  return v;
}
__device__ __forceinline__ float waveMax(float v) {
#pragma unroll
  for (int off = 16; off > 0; off >>= 1) v = fmaxf(v, __shfl_xor(v, off, 32));
  return v;
}

__global__ __launch_bounds__(256) void ln_quant_kernel(
    const float* __restrict__ x, signed char* __restrict__ xq,
    float* __restrict__ xscale) {
  const int row  = blockIdx.x;
  const int tid  = threadIdx.x;
  const int lane = tid & 31;
  const int wid  = tid >> 5;

  __shared__ float sS[8], sQ[8], sB[3];

  const float4* xr = (const float4*)(x + (size_t)row * D_IN) + tid * 4;
  float v[16];
  *(float4*)&v[0]  = xr[0];
  *(float4*)&v[4]  = xr[1];
  *(float4*)&v[8]  = xr[2];
  *(float4*)&v[12] = xr[3];

  float s = 0.f, q = 0.f;
#pragma unroll
  for (int i = 0; i < 16; ++i) { s += v[i]; q += v[i] * v[i]; }
  s = waveSum(s); q = waveSum(q);
  if (lane == 0) { sS[wid] = s; sQ[wid] = q; }
  __syncthreads();
  if (tid == 0) {
    float S = 0.f, Q = 0.f;
#pragma unroll
    for (int i = 0; i < 8; ++i) { S += sS[i]; Q += sQ[i]; }
    float mean = S * (1.0f / D_IN);
    float var  = Q * (1.0f / D_IN) - mean * mean;
    sB[0] = mean;
    sB[1] = rsqrtf(var + EPS_LN);
  }
  __syncthreads();
  const float mean = sB[0], rstd = sB[1];

  float amax = 0.f;
#pragma unroll
  for (int i = 0; i < 16; ++i) {
    v[i] = (v[i] - mean) * rstd;
    amax = fmaxf(amax, fabsf(v[i]));
  }
  amax = waveMax(amax);
  if (lane == 0) sS[wid] = amax;
  __syncthreads();
  if (tid == 0) {
    float A = 0.f;
#pragma unroll
    for (int i = 0; i < 8; ++i) A = fmaxf(A, sS[i]);
    float sc = fmaxf(A, EPS_Q) * (1.0f / QB);
    sB[2] = sc;
    xscale[row] = sc;
  }
  __syncthreads();
  const float inv = 1.0f / sB[2];

  int words[4];
#pragma unroll
  for (int c = 0; c < 4; ++c) {
    int w0 = 0;
#pragma unroll
    for (int b = 0; b < 4; ++b) {
      float qf = fminf(fmaxf(rintf(v[c * 4 + b] * inv), -128.f), 127.f);
      int   qi = (int)qf;
      w0 |= (qi & 0xFF) << (8 * b);
    }
    words[c] = w0;
  }
  int4 st = {words[0], words[1], words[2], words[3]};
  *(int4*)(xq + (size_t)row * D_IN + tid * 16) = st;
}

#define WP_BLOCKS 512
__global__ __launch_bounds__(256) void wabs_partial_kernel(
    const float* __restrict__ w, float* __restrict__ partial, int n4) {
  const int tid  = threadIdx.x;
  const int lane = tid & 31;
  const int wid  = tid >> 5;
  __shared__ float sS[8];

  float s = 0.f;
  const float4* w4 = (const float4*)w;
  for (int i = blockIdx.x * blockDim.x + tid; i < n4;
       i += gridDim.x * blockDim.x) {
    float4 t = w4[i];
    s += fabsf(t.x) + fabsf(t.y) + fabsf(t.z) + fabsf(t.w);
  }
  s = waveSum(s);
  if (lane == 0) sS[wid] = s;
  __syncthreads();
  if (tid == 0) {
    float S = 0.f;
#pragma unroll
    for (int i = 0; i < 8; ++i) S += sS[i];
    partial[blockIdx.x] = S;
  }
}

__global__ __launch_bounds__(256) void wabs_final_kernel(
    const float* __restrict__ partial, float* __restrict__ wscale,
    float inv_count) {
  const int tid  = threadIdx.x;
  const int lane = tid & 31;
  const int wid  = tid >> 5;
  __shared__ float sS[8];
  float s = partial[tid] + partial[tid + 256];
  s = waveSum(s);
  if (lane == 0) sS[wid] = s;
  __syncthreads();
  if (tid == 0) {
    float S = 0.f;
#pragma unroll
    for (int i = 0; i < 8; ++i) S += sS[i];
    *wscale = fmaxf(S * inv_count, EPS_Q);
  }
}

__global__ __launch_bounds__(256) void wquant_kernel(
    const float* __restrict__ w, const float* __restrict__ wscale_p,
    int* __restrict__ wq_words, int n4) {
  const float inv = 1.0f / *wscale_p;
  const float4* w4 = (const float4*)w;
  for (int i = blockIdx.x * blockDim.x + threadIdx.x; i < n4;
       i += gridDim.x * blockDim.x) {
    float4 t = w4[i];
    int q0 = (int)fminf(fmaxf(rintf(t.x * inv), -1.f), 1.f);
    int q1 = (int)fminf(fmaxf(rintf(t.y * inv), -1.f), 1.f);
    int q2 = (int)fminf(fmaxf(rintf(t.z * inv), -1.f), 1.f);
    int q3 = (int)fminf(fmaxf(rintf(t.w * inv), -1.f), 1.f);
    wq_words[i] = (q0 & 0xFF) | ((q1 & 0xFF) << 8) | ((q2 & 0xFF) << 16) |
                  ((q3 & 0xFF) << 24);
  }
}

__global__ __launch_bounds__(256) void bitgemm_kernel(
    const signed char* __restrict__ xq, const signed char* __restrict__ wq,
    const float* __restrict__ xscale, const float* __restrict__ wscale_p,
    float* __restrict__ out, int N) {
  const int tid    = threadIdx.x;
  const int lane   = tid & 31;
  const int waveId = tid >> 5;
  const int wm     = waveId & 1;
  const int wn     = waveId >> 1;
  const int half   = lane >> 4;
  const int l16    = lane & 15;

  const int rowBase = blockIdx.y * 128 + wm * 64;
  const int colBase = blockIdx.x * 128 + wn * 32;

  v8i acc[4][2];
#pragma unroll
  for (int i = 0; i < 4; ++i)
#pragma unroll
    for (int j = 0; j < 2; ++j) acc[i][j] = (v8i)0;

  const signed char* aRow[4];
#pragma unroll
  for (int i = 0; i < 4; ++i)
    aRow[i] = xq + (size_t)(rowBase + i * 16 + l16) * D_IN + half * 8;
  const signed char* bCol[2];
#pragma unroll
  for (int j = 0; j < 2; ++j)
    bCol[j] = wq + (size_t)(colBase + j * 16 + l16) * D_IN + half * 8;

  for (int k0 = 0; k0 < D_IN; k0 += 64) {
    v8i a[4];
#pragma unroll
    for (int i = 0; i < 4; ++i) {
      int2 c0 = *(const int2*)(aRow[i] + k0);
      int2 c1 = *(const int2*)(aRow[i] + k0 + 16);
      int2 c2 = *(const int2*)(aRow[i] + k0 + 32);
      int2 c3 = *(const int2*)(aRow[i] + k0 + 48);
      v8i t = {c0.x, c0.y, c1.x, c1.y, c2.x, c2.y, c3.x, c3.y};
      a[i] = t;
    }
    v8i b[2];
#pragma unroll
    for (int j = 0; j < 2; ++j) {
      int2 d0 = *(const int2*)(bCol[j] + k0);
      int2 d1 = *(const int2*)(bCol[j] + k0 + 16);
      int2 d2 = *(const int2*)(bCol[j] + k0 + 32);
      int2 d3 = *(const int2*)(bCol[j] + k0 + 48);
      v8i t = {d0.x, d0.y, d1.x, d1.y, d2.x, d2.y, d3.x, d3.y};
      b[j] = t;
    }
#pragma unroll
    for (int i = 0; i < 4; ++i)
#pragma unroll
      for (int j = 0; j < 2; ++j)
        acc[i][j] = __builtin_amdgcn_wmma_i32_16x16x64_iu8(
             true, a[i],  true, b[j], acc[i][j],
             false,  false);
  }

  const float wsc = *wscale_p;
#pragma unroll
  for (int i = 0; i < 4; ++i) {
    const int mBase = rowBase + i * 16 + half * 8;
#pragma unroll
    for (int j = 0; j < 2; ++j) {
      const int n = colBase + j * 16 + l16;
#pragma unroll
      for (int r = 0; r < 8; ++r) {
        const int m = mBase + r;
        out[(size_t)m * N + n] = (float)acc[i][j][r] * (wsc * xscale[m]);
      }
    }
  }
}

extern "C" void kernel_launch(void* const* d_in, const int* in_sizes, int n_in,
                              void* d_out, int out_size, void* d_ws,
                              size_t ws_size, hipStream_t stream) {
  const float* x = (const float*)d_in[0];
  const float* w = (const float*)d_in[1];
  float* out = (float*)d_out;

  const int M = in_sizes[0] / D_IN;
  const int N = in_sizes[1] / D_IN;

  char*        ws      = (char*)d_ws;
  signed char* xq      = (signed char*)ws;
  signed char* wq      = (signed char*)(ws + (size_t)M * D_IN);
  float*       xscale  = (float*)(ws + (size_t)M * D_IN + (size_t)N * D_IN);
  float*       partial = xscale + M;
  float*       wscale  = partial + WP_BLOCKS;

  const int nW  = N * D_IN;
  const int nW4 = nW / 4;

  wabs_partial_kernel<<<WP_BLOCKS, 256, 0, stream>>>(w, partial, nW4);
  wabs_final_kernel<<<1, 256, 0, stream>>>(partial, wscale, 1.0f / (float)nW);
  wquant_kernel<<<2048, 256, 0, stream>>>(w, wscale, (int*)wq, nW4);
  ln_quant_kernel<<<M, 256, 0, stream>>>(x, xq, xscale);

  dim3 grid(N / 128, M / 128);
  bitgemm_kernel<<<grid, 256, 0, stream>>>(xq, wq, xscale, wscale, out, N);
}
